// Head_19894288515132
// MI455X (gfx1250) — matrix operand order repaired and checked by reading, not compiled, not run on hardware
//
#include <hip/hip_runtime.h>
#include <stdint.h>

#define BATCH 512
#define TSEQ  256
#define CDIM  384
#define HSD   64
#define BT    (BATCH * TSEQ)          // 131072
#define BT64  ((size_t)BT * HSD)      // 8388608 elements per q/k/v plane

// softmax scale folded into q at projection time, in log2 domain:
// HS^-0.5 * log2(e) = 0.125 * 1.4426950408889634
#define QSCALE 0.18033688011112043f

typedef __attribute__((ext_vector_type(16))) __bf16 v16bf;
typedef __attribute__((ext_vector_type(8)))  float  v8f;

union ABFrag { uint32_t u[8]; v16bf v; };
union CDFrag { float    f[8]; v8f   v; };

// f32 -> bf16 via native fptrunc (selects v_cvt_pk_bf16_f32 on gfx1250)
static __device__ __forceinline__ uint16_t f2bf(float x) {
  __bf16 h = (__bf16)x;
  return __builtin_bit_cast(uint16_t, h);
}
// pack two f32 -> packed bf16 dword (lo in [15:0])
static __device__ __forceinline__ uint32_t pack2(float lo, float hi) {
  union { __bf16 b[2]; uint32_t u; } cv;
  cv.b[0] = (__bf16)lo;
  cv.b[1] = (__bf16)hi;
  return cv.u;
}
static __device__ __forceinline__ float fexp2(float x) {
#if __has_builtin(__builtin_amdgcn_exp2f)
  return __builtin_amdgcn_exp2f(x);     // v_exp_f32 computes 2^x natively
#else
  return exp2f(x);
#endif
}
static __device__ __forceinline__ v8f vzero8() {
  v8f z = {0.f, 0.f, 0.f, 0.f, 0.f, 0.f, 0.f, 0.f};
  return z;
}
static __device__ __forceinline__ v8f wmma_bf16(const ABFrag& a, const ABFrag& b, v8f c) {
  return __builtin_amdgcn_wmma_f32_16x16x32_bf16(false, a.v, false, b.v, (short)0, c,
                                                 false, false);
}
// 16B global -> LDS async copy (ASYNCcnt path). LDS address = low 32 bits of
// the generic shared pointer (flat->LDS aperture truncation).
static __device__ __forceinline__ void async_ld16(void* lds, const void* g) {
  asm volatile("global_load_async_to_lds_b128 %0, %1, off"
               :: "v"((uint32_t)(size_t)lds), "v"((uint64_t)(size_t)g)
               : "memory");
}
static __device__ __forceinline__ void wait_async0() {
  asm volatile("s_wait_asynccnt 0" ::: "memory");
}
static __device__ __forceinline__ void ld_frag_b128(ABFrag& fr, const uint32_t* d) {
  uint4 t0 = *(const uint4*)(d);
  uint4 t1 = *(const uint4*)(d + 4);
  fr.u[0] = t0.x; fr.u[1] = t0.y; fr.u[2] = t0.z; fr.u[3] = t0.w;
  fr.u[4] = t1.x; fr.u[5] = t1.y; fr.u[6] = t1.z; fr.u[7] = t1.w;
}

// ---------------------------------------------------------------------------
// Pass 1: q/k/v projections.  X[131072x384] (fp32) @ W[384x64] (fp32) -> bf16.
// Weights in LDS in fragment-major order [w][ks][half][n][v] (dwords) so each
// B-fragment half is one ds_load_b128.  q is pre-scaled by QSCALE; V plane is
// written to global already in per-batch fragment-major order for pass 2.
// K-loop kept rolled (#pragma unroll 1) so the scheduler multi-buffers the
// fragment loads against the WMMAs instead of serializing on one reg set.
// ---------------------------------------------------------------------------
__global__ __launch_bounds__(256) void proj_qkv(const float* __restrict__ x,
                                                const float* __restrict__ Wq,
                                                const float* __restrict__ Wk,
                                                const float* __restrict__ Wv,
                                                uint16_t* __restrict__ qkv) {
  __shared__ __align__(16) uint16_t wlds[3 * CDIM * HSD];   // 144 KB of 320 KB LDS
  uint32_t* wlds32 = (uint32_t*)wlds;

  const int tid = threadIdx.x;
  for (int idx = tid; idx < 3 * CDIM * HSD; idx += 256) {
    const int w   = idx / (CDIM * HSD);
    const int rem = idx - w * (CDIM * HSD);
    const int k   = rem >> 6;
    const int n   = rem & 63;
    const float* wp = (w == 0) ? Wq : ((w == 1) ? Wk : Wv);
    // fragment-major: dword = (((w*12 + k/32)*2 + (k%32)/16)*64 + n)*8 + (k%16)/2
    const int dw = (((w * 12 + (k >> 5)) * 2 + ((k >> 4) & 1)) * 64 + n) * 8 + ((k & 15) >> 1);
    wlds[dw * 2 + (k & 1)] = f2bf(wp[k * HSD + n]);
  }
  __syncthreads();

  const int wave  = tid >> 5;
  const int lane  = tid & 31;
  const int lrow  = lane & 15;
  const int half  = lane >> 4;
  const int kgrp8 = half * 8;          // unused; A-fragment K base within 32-chunk is half*16
  const int row0  = blockIdx.x * 128 + wave * 16;

  CDFrag acc[3][4];
#pragma unroll
  for (int w = 0; w < 3; w++)
#pragma unroll
    for (int f = 0; f < 4; f++) acc[w][f].v = vzero8();

  const float* xrow = x + (size_t)(row0 + lrow) * CDIM;

#pragma unroll 1
  for (int ks = 0; ks < CDIM / 32; ks++) {
    // A fragment: 16x32 bf16 of x (one contiguous 16-float run per lane)
    ABFrag a;
    const int kb = ks * 32 + half * 16;
    float4 f0 = *(const float4*)(xrow + kb);
    float4 f1 = *(const float4*)(xrow + kb + 4);
    float4 f2 = *(const float4*)(xrow + kb + 8);
    float4 f3 = *(const float4*)(xrow + kb + 12);
    a.u[0] = pack2(f0.x, f0.y); a.u[1] = pack2(f0.z, f0.w);
    a.u[2] = pack2(f1.x, f1.y); a.u[3] = pack2(f1.z, f1.w);
    a.u[4] = pack2(f2.x, f2.y); a.u[5] = pack2(f2.z, f2.w);
    a.u[6] = pack2(f3.x, f3.y); a.u[7] = pack2(f3.z, f3.w);
#pragma unroll
    for (int w = 0; w < 3; w++) {
#pragma unroll
      for (int f = 0; f < 4; f++) {
        ABFrag b;
        ld_frag_b128(b, wlds32 + ((((w * 12 + ks) * 2 + half) * 64 + (f * 16 + lrow)) * 8));
        acc[w][f].v = wmma_bf16(a, b, acc[w][f].v);
      }
    }
  }

  // q plane (pre-scaled) and k plane: row-major bf16
#pragma unroll
  for (int f = 0; f < 4; f++)
#pragma unroll
    for (int r = 0; r < 8; r++) {
      const int row = row0 + r + 8 * half;
      const int col = f * 16 + lrow;
      qkv[(size_t)row * HSD + col]        = f2bf(acc[0][f].f[r] * QSCALE);
      qkv[BT64 + (size_t)row * HSD + col] = f2bf(acc[1][f].f[r]);
    }
  // v plane: per-batch fragment-major so pass 2 can straight-copy into LDS
#pragma unroll
  for (int f = 0; f < 4; f++)
#pragma unroll
    for (int r = 0; r < 8; r++) {
      const int row = row0 + r + 8 * half;
      const int col = f * 16 + lrow;
      const int bb  = row >> 8;
      const int t   = row & 255;      // key index = WMMA K-dim of the P@V GEMM
      const int dw  = (((t >> 5) * 2 + ((t >> 4) & 1)) * 64 + col) * 8 + ((t & 15) >> 1);
      qkv[(size_t)2 * BT64 + (size_t)bb * (TSEQ * HSD) + dw * 2 + (t & 1)] =
          f2bf(acc[2][f].f[r]);
    }
}

// ---------------------------------------------------------------------------
// Pass 2: fused causal flash attention, one block per (batch, 128-query tile).
// K/V staged to LDS with global_load_async_to_lds_b128 (ASYNCcnt), per-wave
// online softmax (log2 domain, single v_exp_f32 per probability) over 32-key
// tiles, causal tiles skipped by trip count.
// ---------------------------------------------------------------------------
__global__ __launch_bounds__(256) void attn_fused(const uint16_t* __restrict__ qkv,
                                                  float* __restrict__ out) {
  __shared__ __align__(16) uint16_t klds[TSEQ * HSD];  // row-major [key][hs]  (32 KB)
  __shared__ __align__(16) uint16_t vlds[TSEQ * HSD];  // fragment-major       (32 KB)
  __shared__ __align__(16) uint16_t plds[8][16 * 32];  // per-wave P bounce    ( 8 KB)
  const uint32_t* klds32 = (const uint32_t*)klds;
  const uint32_t* vlds32 = (const uint32_t*)vlds;

  const int tid   = threadIdx.x;
  const int b     = blockIdx.x >> 1;
  const int qtile = blockIdx.x & 1;

  const uint16_t* qbase = qkv + (size_t)b * TSEQ * HSD;
  const uint16_t* kbase = qkv + BT64 + (size_t)b * TSEQ * HSD;
  const uint16_t* vbase = qkv + 2 * BT64 + (size_t)b * TSEQ * HSD;

  // async-stage K and V (32 KB each): 16 B per lane per issue
#pragma unroll
  for (int it = 0; it < 8; ++it) {
    const int off = (it * 256 + tid) * 16;
    async_ld16((char*)klds + off, (const char*)kbase + off);
    async_ld16((char*)vlds + off, (const char*)vbase + off);
  }
  wait_async0();
  __syncthreads();   // only block-wide barrier; before divergent loops

  const int wave   = tid >> 5;
  const int lane   = tid & 31;
  const int lrow   = lane & 15;
  const int half   = lane >> 4;
  const int kgrp8  = half * 8;
  const int kgrp16 = half * 16;
  const int q0     = qtile * 128 + wave * 16;   // wave's query-row base in [0,256)

  // Q A-fragments: 16x64 split as two 16x32
  ABFrag qa[2];
  const uint16_t* qrow = qbase + (size_t)(q0 + lrow) * HSD;
#pragma unroll
  for (int h = 0; h < 2; h++) {
    const int kb = h * 32 + kgrp16;
    uint4 t0 = *(const uint4*)(qrow + kb);
    uint4 t1 = *(const uint4*)(qrow + kb + 8);
    qa[h].u[0] = t0.x; qa[h].u[1] = t0.y; qa[h].u[2] = t0.z; qa[h].u[3] = t0.w;
    qa[h].u[4] = t1.x; qa[h].u[5] = t1.y; qa[h].u[6] = t1.z; qa[h].u[7] = t1.w;
  }

  CDFrag o[4];
#pragma unroll
  for (int f = 0; f < 4; f++) o[f].v = vzero8();
  float mrow[8], lsum[8];
#pragma unroll
  for (int r = 0; r < 8; r++) { mrow[r] = -1e30f; lsum[r] = 0.f; }

  const int nkt = ((q0 + 15) >> 5) + 1;      // causal: skip fully-masked tiles
#pragma unroll 1
  for (int kt = 0; kt < nkt; kt++) {
    // S = Q @ K^T for 32 keys -> two 16x16 f32 fragments (log2-domain logits)
    CDFrag s[2];
    s[0].v = vzero8(); s[1].v = vzero8();
#pragma unroll
    for (int h = 0; h < 2; h++) {
#pragma unroll
      for (int c = 0; c < 2; c++) {
        ABFrag bfr;
        const int key = kt * 32 + c * 16 + lrow;
        ld_frag_b128(bfr, klds32 + key * 32 + (h * 32 + kgrp16) / 2);
        s[c].v = wmma_bf16(qa[h], bfr, s[c].v);
      }
    }

    // online softmax (row = r + 8*half, replicated over the 16 lanes/row)
#pragma unroll
    for (int r = 0; r < 8; r++) {
      const int grow = q0 + r + 8 * half;
      const int col0 = kt * 32 + lrow;
      float s0 = s[0].f[r];  if (col0      > grow) s0 = -1e30f;
      float s1 = s[1].f[r];  if (col0 + 16 > grow) s1 = -1e30f;
      float rm = fmaxf(s0, s1);
#pragma unroll
      for (int off = 8; off >= 1; off >>= 1) rm = fmaxf(rm, __shfl_xor(rm, off, 32));
      const float mnew = fmaxf(mrow[r], rm);
      const float resc = fexp2(mrow[r] - mnew);
      const float p0   = fexp2(s0 - mnew);
      const float p1   = fexp2(s1 - mnew);
      float rs = p0 + p1;
#pragma unroll
      for (int off = 8; off >= 1; off >>= 1) rs += __shfl_xor(rs, off, 32);
      lsum[r] = lsum[r] * resc + rs;
      mrow[r] = mnew;
#pragma unroll
      for (int f = 0; f < 4; f++) o[f].f[r] *= resc;
      const int prow = r + 8 * half;
      plds[wave][prow * 32 + lrow]      = f2bf(p0);
      plds[wave][prow * 32 + 16 + lrow] = f2bf(p1);
    }

    // C-layout -> A-layout bounce through wave-private LDS
    asm volatile("s_wait_dscnt 0" ::: "memory");
    ABFrag pa;
    ld_frag_b128(pa, (const uint32_t*)&plds[wave][0] + lrow * 16 + kgrp16 / 2);

    // O += P(16x32) @ V(32x64); V fragments are contiguous (fragment-major)
#pragma unroll
    for (int f = 0; f < 4; f++) {
      ABFrag bfr;
      ld_frag_b128(bfr, vlds32 + ((kt * 2 + half) * 64 + (f * 16 + lrow)) * 8);
      o[f].v = wmma_bf16(pa, bfr, o[f].v);
    }
  }

  float* obase = out + (size_t)b * TSEQ * HSD;
#pragma unroll
  for (int r = 0; r < 8; r++) {
    const float inv  = 1.0f / lsum[r];
    const int   grow = q0 + r + 8 * half;
#pragma unroll
    for (int f = 0; f < 4; f++)
      obase[(size_t)grow * HSD + f * 16 + lrow] = o[f].f[r] * inv;
  }
}

// ---------------------------------------------------------------------------
extern "C" void kernel_launch(void* const* d_in, const int* in_sizes, int n_in,
                              void* d_out, int out_size, void* d_ws, size_t ws_size,
                              hipStream_t stream) {
  (void)in_sizes; (void)n_in; (void)out_size; (void)ws_size;
  const float* x  = (const float*)d_in[0];
  const float* Wq = (const float*)d_in[1];
  const float* Wk = (const float*)d_in[2];
  const float* Wv = (const float*)d_in[3];
  uint16_t* qkv = (uint16_t*)d_ws;        // 3 * 131072 * 64 bf16 = 50.3 MB scratch
  float* out = (float*)d_out;

  proj_qkv<<<BT / 128, 256, 0, stream>>>(x, Wq, Wk, Wv, qkv);
  attn_fused<<<BATCH * 2, 256, 0, stream>>>(qkv, out);
}
